// PEdgeLSTMTrain_35021163331778
// MI455X (gfx1250) — matrix operand order repaired and checked by reading, not compiled, not run on hardware
//
#include <hip/hip_runtime.h>
#include <hip/hip_bf16.h>

typedef __attribute__((ext_vector_type(16))) _Float16 v16h;
typedef __attribute__((ext_vector_type(8)))  float    v8f;

#define N0v 100000
#define N1v 20000
#define Ev  320000
#define Tv  8
#define DIN 128
#define Hv  64
#define EIN 15
#define FL  16    // F_LSTM = E_IN + 1
#define G4H 256   // 4*H
#define KC  192   // D_IN + H (A-operand row stride, halfs)
#define CO  16

#define LOG2E 1.442695041f

// ---------------- branchless activations (single-instr transcendentals) -----
__device__ __forceinline__ float fast_sigmoid(float x) {
  // 1 / (1 + 2^(-x*log2e)) : v_exp_f32 + v_rcp_f32, no branches
  return __builtin_amdgcn_rcpf(1.f + __builtin_amdgcn_exp2f(-x * LOG2E));
}
__device__ __forceinline__ float fast_tanh(float x) {
#if __has_builtin(__builtin_amdgcn_tanhf)
  return __builtin_amdgcn_tanhf(x);          // v_tanh_f32 on gfx1250
#elif __has_builtin(__builtin_amdgcn_tanh_f32)
  return __builtin_amdgcn_tanh_f32(x);
#else
  // 2*sigmoid(2x) - 1, branchless
  return 2.f * __builtin_amdgcn_rcpf(1.f + __builtin_amdgcn_exp2f(-2.f * x * LOG2E)) - 1.f;
#endif
}

// ---------------- WMMA operand loaders (wave32, 16x16x32 f16) ----------------
// A (16x32 f16): lane L holds row M=L&15; per-half K:
//   k = (hi/8)*16 + ((L>=16)?8:0) + (hi%8)
__device__ __forceinline__ v16h load_a16(const _Float16* __restrict__ s, int lda, int kbase) {
  const int lane = threadIdx.x & 31;
  const int m    = lane & 15;
  const int koff = (lane & 16) ? 8 : 0;
  const _Float16* row = s + m * lda + kbase + koff;
  v16h a;
#pragma unroll
  for (int hi = 0; hi < 16; ++hi) {
    int k = ((hi >> 3) << 4) + (hi & 7);
    a[hi] = row[k];
  }
  return a;
}

// B (32x16 f16, row-major K x N in LDS): lane L holds column N=L&15,
// K = (hi/8)*16 + ((L>=16)?8:0) + (hi%8)
__device__ __forceinline__ v16h load_b16(const _Float16* __restrict__ s, int ldw, int kbase, int nbase) {
  const int lane = threadIdx.x & 31;
  const int n  = (lane & 15) + nbase;
  const int k0 = kbase + ((lane & 16) ? 8 : 0);
  v16h b;
#pragma unroll
  for (int hi = 0; hi < 16; ++hi) b[hi] = s[(k0 + ((hi >> 3) << 4) + (hi & 7)) * ldw + n];
  return b;
}

__global__ void zero_f32(float* __restrict__ p, int n) {
  int i = blockIdx.x * blockDim.x + threadIdx.x;
  if (i < n) p[i] = 0.f;
}

// =============== Kernel 1: per-edge LSTM + message + scatter-sum =============
__global__ __launch_bounds__(256) void edge_lstm_msg(
    const float* __restrict__ nodef, const float* __restrict__ ef,
    const float* __restrict__ st,    const float* __restrict__ pw,
    const float* __restrict__ Wih,   const float* __restrict__ Whh,
    const float* __restrict__ bL,    const float* __restrict__ Wo,
    const float* __restrict__ bo,    const int* __restrict__ elen,
    const int* __restrict__ src,     const int* __restrict__ dst,
    float* __restrict__ agg)
{
  extern __shared__ char smem_raw[];
  _Float16* sW  = (_Float16*)smem_raw;      // [96][256]  combined [Wih;Whh;pad]
  _Float16* sWo = sW  + 96 * 256;           // [192][64]  edge_out_W
  _Float16* sA  = sWo + 192 * 64;           // [16][192]  A staging
  float* sGat = (float*)(sA + 16 * KC);     // [16][256]  gates
  float* sH   = sGat + 16 * 256;            // [16][64]
  float* sC   = sH   + 16 * 64;
  float* sLa  = sC   + 16 * 64;             // last_h
  float* sHS  = sLa  + 16 * 64;             // [16][128]  gathered h_src
  float* sRed = sHS  + 16 * 128;            // [16][16]   dot-product partials
  float* sB   = sRed + 16 * 16;             // [256]      lstm bias

  const int tid  = threadIdx.x;
  const int wave = tid >> 5;
  const int lane = tid & 31;
  const int rr   = tid >> 4;                // row 0..15 owned by this thread
  const int cc   = tid & 15;                // sub-column group

  // ---- stage weights (f32 -> f16) into LDS, once per block ----
  for (int i = tid; i < 96 * 256; i += 256) {
    int r = i >> 8, c = i & 255;
    float v = 0.f;
    if (r < FL)           v = Wih[r * G4H + c];
    else if (r < FL + Hv) v = Whh[(r - FL) * G4H + c];
    sW[i] = (_Float16)v;
  }
  for (int i = tid; i < 192 * 64; i += 256) sWo[i] = (_Float16)Wo[i];
  sB[tid] = bL[tid];
  __syncthreads();

  // pre-load this wave's LSTM B tiles (N columns [32*wave, 32*wave+32))
  v16h Bw[2][3];
#pragma unroll
  for (int sub = 0; sub < 2; ++sub)
#pragma unroll
    for (int kt = 0; kt < 3; ++kt)
      Bw[sub][kt] = load_b16(sW, G4H, kt * 32, wave * 32 + sub * 16);

  for (int tile = blockIdx.x; tile < Ev / 16; tile += gridDim.x) {
    const int e0 = tile * 16;
    {
      int ntile = tile + gridDim.x;
      if (ntile < Ev / 16)
        __builtin_prefetch(&ef[(size_t)(ntile * 16) * Tv * EIN], 0, 1);
    }
    const int len_r = elen[e0 + rr];         // per-thread row metadata, hoisted
    const int src_r = src[e0 + rr];

    for (int i = tid; i < 16 * 64; i += 256) { sH[i] = 0.f; sC[i] = 0.f; sLa[i] = 0.f; }
    sA[rr * KC + 80 + cc] = (_Float16)0.f;   // zero K pad cols 80..95
    __syncthreads();

    // ---------------- 8 LSTM timesteps ----------------
    for (int t = 0; t < Tv; ++t) {
      { // x_t -> sA[:, 0:16]   (15 edge feats + appended time)
        int e = e0 + rr;
        float v = (cc < EIN) ? ef[((size_t)e * Tv + t) * EIN + cc]
                             : st[(size_t)e * Tv + t];
        sA[rr * KC + cc] = (_Float16)v;
      }
#pragma unroll
      for (int u = 0; u < 4; ++u) {          // h -> sA[:, 16:80]
        int j = cc * 4 + u;
        sA[rr * KC + FL + j] = (_Float16)sH[rr * 64 + j];
      }
      __syncthreads();

      v8f acc0 = {}; v8f acc1 = {};
#pragma unroll
      for (int kt = 0; kt < 3; ++kt) {
        v16h a = load_a16(sA, KC, kt * 32);
        acc0 = __builtin_amdgcn_wmma_f32_16x16x32_f16(false, a, false, Bw[0][kt], (short)0, acc0, false, false);
        acc1 = __builtin_amdgcn_wmma_f32_16x16x32_f16(false, a, false, Bw[1][kt], (short)0, acc1, false, false);
      }
      { // write gates (+bias) to LDS
        int rowoff = (lane & 16) ? 8 : 0;
        int c0 = wave * 32 + (lane & 15);
        float b0 = sB[c0], b1 = sB[c0 + 16];
#pragma unroll
        for (int r2 = 0; r2 < 8; ++r2) {
          int row = r2 + rowoff;
          sGat[row * 256 + c0]      = acc0[r2] + b0;
          sGat[row * 256 + c0 + 16] = acc1[r2] + b1;
        }
      }
      __syncthreads();

#pragma unroll
      for (int u = 0; u < 4; ++u) {          // LSTM cell update (branchless)
        int j = cc * 4 + u;
        float gi = sGat[rr * 256 + j];
        float gf = sGat[rr * 256 + 64 + j];
        float gg = sGat[rr * 256 + 128 + j];
        float go = sGat[rr * 256 + 192 + j];
        float c = fast_sigmoid(gf) * sC[rr * 64 + j] + fast_sigmoid(gi) * fast_tanh(gg);
        float h = fast_sigmoid(go) * fast_tanh(c);
        sC[rr * 64 + j] = c; sH[rr * 64 + j] = h;
        if (t < len_r) sLa[rr * 64 + j] = h;
      }
      __syncthreads();
    }

    // ------------- message: relu([h_proj | e_enc] @ Wo + bo) -------------
#pragma unroll
    for (int u = 0; u < 8; ++u) {            // gather h_src rows (coalesced per row)
      int k = cc * 8 + u;
      sHS[rr * 128 + k] = nodef[(size_t)src_r * DIN + k];
    }
    __syncthreads();
    { // rank-1 projection dot(h_src, p): partials
      int r = tid & 15, chunk = tid >> 4;
      float p = 0.f;
#pragma unroll
      for (int u = 0; u < 8; ++u) {
        int k = chunk * 8 + u;
        p += sHS[r * 128 + k] * pw[k];
      }
      sRed[r * 16 + chunk] = p;
    }
    __syncthreads();
    if (tid < 16) {
      float d = 0.f;
#pragma unroll
      for (int u = 0; u < 16; ++u) d += sRed[tid * 16 + u];
      sRed[tid * 16] = d;
    }
    __syncthreads();
#pragma unroll
    for (int u = 0; u < 8; ++u) {            // h_proj -> sA[:, 0:128]
      int k = cc * 8 + u;
      sA[rr * KC + k] = (_Float16)(sHS[rr * 128 + k] - sRed[rr * 16] * pw[k]);
    }
#pragma unroll
    for (int u = 0; u < 4; ++u) {            // e_enc -> sA[:, 128:192]
      int j = cc * 4 + u;
      sA[rr * KC + DIN + j] = (_Float16)sLa[rr * 64 + j];
    }
    __syncthreads();

    if (wave < 4) {                          // 16x64 output, K=192
      v8f acc = {};
#pragma unroll
      for (int kt = 0; kt < 6; ++kt) {
        v16h a = load_a16(sA, KC, kt * 32);
        v16h b = load_b16(sWo, Hv, kt * 32, wave * 16);
        acc = __builtin_amdgcn_wmma_f32_16x16x32_f16(false, a, false, b, (short)0, acc, false, false);
      }
      int rowoff = (lane & 16) ? 8 : 0;
      int ncol = wave * 16 + (lane & 15);
      float bias = bo[ncol];
#pragma unroll
      for (int r2 = 0; r2 < 8; ++r2) {
        int row = r2 + rowoff;
        float m = acc[r2] + bias;
        m = m > 0.f ? m : 0.f;
        atomicAdd(&agg[(size_t)dst[e0 + row] * Hv + ncol], m);
      }
    }
    __syncthreads();
  }
}

// ===================== Kernel 2: node update + classifier ====================
__global__ __launch_bounds__(256) void node_update(
    const float* __restrict__ nodef, const float* __restrict__ subg,
    const float* __restrict__ Wo,    const float* __restrict__ bo,
    const float* __restrict__ Wn,    const float* __restrict__ bn,
    const float* __restrict__ Wfc,   const float* __restrict__ bfc,
    const int* __restrict__ lnid,    const float* __restrict__ agg,
    float* __restrict__ out)
{
  extern __shared__ char smem_raw[];
  _Float16* sWo  = (_Float16*)smem_raw;     // [192][64]
  _Float16* sWn  = sWo + 192 * 64;          // [192][64]
  _Float16* sFc  = sWn + 192 * 64;          // [64][16]
  _Float16* sA   = sFc + 64 * 16;           // [16][192]
  _Float16* sAct = sA  + 16 * KC;           // [16][64]
  float*    sTmp = (float*)(sAct + 16 * 64);// [16][64]

  const int tid  = threadIdx.x;
  const int wave = tid >> 5;
  const int lane = tid & 31;
  const int rr   = tid >> 4;
  const int cc   = tid & 15;

  for (int i = tid; i < 192 * 64; i += 256) { sWo[i] = (_Float16)Wo[i]; sWn[i] = (_Float16)Wn[i]; }
  for (int i = tid; i < 64 * 16; i += 256) sFc[i] = (_Float16)Wfc[i];
  __syncthreads();

  const int n0 = blockIdx.x * 16;
  const int nid_r = lnid[n0 + rr];
#pragma unroll
  for (int u = 0; u < 8; ++u) {             // self_h -> sA[:, 0:128]
    int k = cc * 8 + u;
    sA[rr * KC + k] = (_Float16)nodef[(size_t)nid_r * DIN + k];
  }
#pragma unroll
  for (int u = 0; u < 4; ++u) {             // zero cols 128:192
    int j = cc * 4 + u;
    sA[rr * KC + DIN + j] = (_Float16)0.f;
  }
  __syncthreads();

  if (wave < 4) {                            // self_h_tmp (K=128 effective)
    v8f acc = {};
#pragma unroll
    for (int kt = 0; kt < 4; ++kt) {
      v16h a = load_a16(sA, KC, kt * 32);
      v16h b = load_b16(sWo, Hv, kt * 32, wave * 16);
      acc = __builtin_amdgcn_wmma_f32_16x16x32_f16(false, a, false, b, (short)0, acc, false, false);
    }
    int rowoff = (lane & 16) ? 8 : 0;
    int ncol = wave * 16 + (lane & 15);
    float bias = bo[ncol];
#pragma unroll
    for (int r2 = 0; r2 < 8; ++r2)
      sTmp[(r2 + rowoff) * 64 + ncol] = acc[r2] + bias;
  }
  __syncthreads();

  { // hu -> sA[:, 128:192]
    float sn = subg[n0 + rr];
#pragma unroll
    for (int u = 0; u < 4; ++u) {
      int j = cc * 4 + u;
      float hu = (agg[(size_t)(n0 + rr) * Hv + j] - sTmp[rr * 64 + j]) * sn;
      sA[rr * KC + DIN + j] = (_Float16)hu;
    }
  }
  __syncthreads();

  if (wave < 4) {                            // act = relu([self_h|hu] @ Wn + bn)
    v8f acc = {};
#pragma unroll
    for (int kt = 0; kt < 6; ++kt) {
      v16h a = load_a16(sA, KC, kt * 32);
      v16h b = load_b16(sWn, Hv, kt * 32, wave * 16);
      acc = __builtin_amdgcn_wmma_f32_16x16x32_f16(false, a, false, b, (short)0, acc, false, false);
    }
    int rowoff = (lane & 16) ? 8 : 0;
    int ncol = wave * 16 + (lane & 15);
    float bias = bn[ncol];
#pragma unroll
    for (int r2 = 0; r2 < 8; ++r2) {
      float v = acc[r2] + bias;
      v = v > 0.f ? v : 0.f;
      sAct[(r2 + rowoff) * 64 + ncol] = (_Float16)v;
    }
  }
  __syncthreads();

  if (wave == 0) {                           // out = act @ fc_W + fc_b  (N=16)
    v8f acc = {};
#pragma unroll
    for (int kt = 0; kt < 2; ++kt) {
      v16h a = load_a16(sAct, Hv, kt * 32);
      v16h b = load_b16(sFc, CO, kt * 32, 0);
      acc = __builtin_amdgcn_wmma_f32_16x16x32_f16(false, a, false, b, (short)0, acc, false, false);
    }
    int rowoff = (lane & 16) ? 8 : 0;
    int ncol = lane & 15;
    float bias = bfc[ncol];
#pragma unroll
    for (int r2 = 0; r2 < 8; ++r2)
      out[(size_t)(n0 + r2 + rowoff) * CO + ncol] = acc[r2] + bias;
  }
}

extern "C" void kernel_launch(void* const* d_in, const int* in_sizes, int n_in,
                              void* d_out, int out_size, void* d_ws, size_t ws_size,
                              hipStream_t stream) {
  const float* nodef = (const float*)d_in[0];
  const float* ef    = (const float*)d_in[1];
  const float* st    = (const float*)d_in[2];
  const float* subg  = (const float*)d_in[3];
  const float* pw    = (const float*)d_in[4];
  const float* Wih   = (const float*)d_in[5];
  const float* Whh   = (const float*)d_in[6];
  const float* bL    = (const float*)d_in[7];
  const float* Wo    = (const float*)d_in[8];
  const float* bo    = (const float*)d_in[9];
  const float* Wn    = (const float*)d_in[10];
  const float* bn    = (const float*)d_in[11];
  const float* Wfc   = (const float*)d_in[12];
  const float* bfc   = (const float*)d_in[13];
  const int* elen = (const int*)d_in[14];
  const int* src  = (const int*)d_in[15];
  const int* dst  = (const int*)d_in[16];
  const int* lnid = (const int*)d_in[17];
  float* agg = (float*)d_ws;                 // [N1][64] f32 accumulator
  float* out = (float*)d_out;

  const int nz = N1v * Hv;
  zero_f32<<<(nz + 255) / 256, 256, 0, stream>>>(agg, nz);

  const size_t smem_e =
      (size_t)(96 * 256 + 192 * 64 + 16 * KC) * 2 +                      // f16
      (size_t)(16 * 256 + 3 * 16 * 64 + 16 * 128 + 16 * 16 + 256) * 4;   // f32
  edge_lstm_msg<<<2048, 256, smem_e, stream>>>(nodef, ef, st, pw, Wih, Whh,
                                               bL, Wo, bo, elen, src, dst, agg);

  const size_t smem_n =
      (size_t)(2 * 192 * 64 + 64 * 16 + 16 * KC + 16 * 64) * 2 +         // f16
      (size_t)(16 * 64) * 4;                                             // f32
  node_update<<<N1v / 16, 256, smem_n, stream>>>(nodef, subg, Wo, bo, Wn, bn,
                                                 Wfc, bfc, lnid, agg, out);
}
